// RNNNet_32220844654811
// MI455X (gfx1250) — matrix operand order repaired and checked by reading, not compiled, not run on hardware
//
#include <hip/hip_runtime.h>
#include <hip/hip_bf16.h>

#define B_SZ   128
#define T_SZ   256
#define IN_SZ  4096
#define H_SZ   512
#define OUT_SZ 4096

typedef __attribute__((ext_vector_type(16))) __bf16 v16bf;
typedef __attribute__((ext_vector_type(8)))  __bf16 v8bf;
typedef __attribute__((ext_vector_type(8)))  float  v8f;

// ---- workspace layout (bytes) ----
#define OFF_WIHT  0ull                                   // f32  [4096][512]  W_ih^T
#define OFF_WHH   (OFF_WIHT + (size_t)IN_SZ*H_SZ*4)      // bf16 [512][512]   W_hh (row-major)
#define OFF_WFC   (OFF_WHH  + (size_t)H_SZ*H_SZ*2)       // bf16 [4096][512]  W_fc (row-major)
#define OFF_BIAS  (OFF_WFC  + (size_t)OUT_SZ*H_SZ*2)     // f32  [512]        b_ih+b_hh
#define OFF_HLAST (OFF_BIAS + (size_t)H_SZ*4)            // bf16 [128][512]   h_last

#define HPAD 520       // padded LDS row stride (bf16): 2-pass conflict-free b128 reads

#define KB_STREAM 5    // K-blocks streamed from L2 each step (per tile)
#define KB_REG    11   // K-blocks held in registers (per tile): 2*11*8 = 176 VGPRs

#if __has_builtin(__builtin_amdgcn_tanhf)
#define FAST_TANH(x) __builtin_amdgcn_tanhf(x)
#else
#define FAST_TANH(x) tanhf(x)
#endif

static __device__ __forceinline__ v16bf join16(v8bf lo, v8bf hi) {
  return __builtin_shufflevector(lo, hi, 0,1,2,3,4,5,6,7,8,9,10,11,12,13,14,15);
}

// ---------------- prep: transpose W_ih, bf16-convert W_hh/W_fc, fold biases ----
__global__ __launch_bounds__(256) void prep_kernel(
    const float* __restrict__ Wih, const float* __restrict__ bih,
    const float* __restrict__ Whh, const float* __restrict__ bhh,
    const float* __restrict__ Wfc,
    float* __restrict__ WihT, __bf16* __restrict__ whh_bf,
    __bf16* __restrict__ wfc_bf, float* __restrict__ biasc) {
  int i = blockIdx.x * blockDim.x + threadIdx.x;
  if (i < IN_SZ * H_SZ) {
    int h = i >> 12;          // i / 4096  (W_ih is (H, IN) row-major)
    int c = i & (IN_SZ - 1);  // i % 4096
    WihT[(size_t)c * H_SZ + h] = Wih[i];
    wfc_bf[i] = (__bf16)Wfc[i];
  }
  if (i < H_SZ * H_SZ) whh_bf[i] = (__bf16)Whh[i];
  if (i < H_SZ)        biasc[i]  = bih[i] + bhh[i];
}

// ---------------- recurrence ----------------
// 8 blocks (batch tiles of 16), 512 threads = 16 waves, 2 N-tiles per wave.
// Each A fragment (h, from LDS) feeds two WMMAs (ds:wmma = 1:1).
// 11/16 K-blocks of W_hh live in registers for all 256 steps; the other 5 are
// streamed from L2 each step via GLOBAL loads with immediate offsets. LICM is
// defeated by an opaque 32-bit zero offset added ONCE per step to the base
// pointers (keeps the global address space -> global_load_b128, not flat).
__global__ __launch_bounds__(512) void rnn_rec_kernel(
    const int* __restrict__ x, const float* __restrict__ WihT,
    const __bf16* __restrict__ whh, const float* __restrict__ biasc,
    __bf16* __restrict__ hlast) {
  __shared__ __bf16 hl[16 * HPAD];
  __shared__ int xs[16];

  const int tid    = threadIdx.x;
  const int lane   = tid & 31;
  const int wv     = tid >> 5;            // wave 0..15 -> N-tiles 2wv, 2wv+1
  const int b0     = blockIdx.x * 16;
  const int Mlo    = lane & 15;
  const int hiHalf = (lane >> 4) & 1;

  for (int i = tid; i < 16 * HPAD; i += 512) hl[i] = (__bf16)0.0f;

  const int n0 = wv * 32 + Mlo;           // lane's column in tile 2wv
  const int n1 = n0 + 16;                 // lane's column in tile 2wv+1
  const float bias0 = biasc[n0];
  const float bias1 = biasc[n1];

  const int aCol = hiHalf ? 16 : 0;       // A fragment K sub-offset
  const int bK   = hiHalf ? 16 : 0;       // B fragment K sub-offset

  // B[k][n] = W_hh[n][k] -> row n of row-major W_hh, contiguous in k
  const __bf16* wrow0 = whh + (size_t)n0 * H_SZ;
  const __bf16* wrow1 = whh + (size_t)n1 * H_SZ;

  // register-resident W: K-blocks KB_STREAM..15 (176 VGPRs total)
  v16bf wf0[KB_REG], wf1[KB_REG];
#pragma unroll
  for (int i = 0; i < KB_REG; ++i) {
    const int k0 = (KB_STREAM + i) * 32 + bK;
    wf0[i] = join16(*(const v8bf*)(wrow0 + k0), *(const v8bf*)(wrow0 + k0 + 8));
    wf1[i] = join16(*(const v8bf*)(wrow1 + k0), *(const v8bf*)(wrow1 + k0 + 8));
  }

  // lane-constant streamed-load base pointers
  const __bf16* sbase0 = wrow0 + bK;
  const __bf16* sbase1 = wrow1 + bK;

  for (int t = 0; t < T_SZ; ++t) {
    if (tid < 16) xs[tid] = x[(size_t)(b0 + tid) * T_SZ + t];
    __syncthreads();   // xs ready; prev-step h writes visible

    // snapshot indices (registers) so xs can be rewritten next step
    int xi[8];
#pragma unroll
    for (int r = 0; r < 8; ++r) xi[r] = xs[r + hiHalf * 8];

    // opaque zero offset, applied ONCE per step: loads stay in the loop (no
    // hoist/spill), base pointers stay provably global (global_load_b128),
    // and each kb displacement folds into the instruction immediate.
    int zoff = 0;
    asm volatile("" : "+v"(zoff));
    const __bf16* s0 = sbase0 + zoff;
    const __bf16* s1 = sbase1 + zoff;

    v8f acc0 = {0,0,0,0,0,0,0,0};
    v8f acc1 = {0,0,0,0,0,0,0,0};

    // --- streamed K-blocks first: loads issue early, drain under reg-WMMAs ---
#pragma unroll
    for (int kb = 0; kb < KB_STREAM; ++kb) {
      v16bf bm0 = join16(*(const v8bf*)(s0 + kb * 32), *(const v8bf*)(s0 + kb * 32 + 8));
      v16bf bm1 = join16(*(const v8bf*)(s1 + kb * 32), *(const v8bf*)(s1 + kb * 32 + 8));
      const __bf16* ap = &hl[Mlo * HPAD + kb * 32 + aCol];
      v16bf a = join16(*(const v8bf*)(ap), *(const v8bf*)(ap + 8));
      acc0 = __builtin_amdgcn_wmma_f32_16x16x32_bf16(false, a, false, bm0,
                                                     (short)0, acc0, false, false);
      acc1 = __builtin_amdgcn_wmma_f32_16x16x32_bf16(false, a, false, bm1,
                                                     (short)0, acc1, false, false);
    }
    // --- register-resident K-blocks ---
#pragma unroll
    for (int i = 0; i < KB_REG; ++i) {
      const __bf16* ap = &hl[Mlo * HPAD + (KB_STREAM + i) * 32 + aCol];
      v16bf a = join16(*(const v8bf*)(ap), *(const v8bf*)(ap + 8));
      acc0 = __builtin_amdgcn_wmma_f32_16x16x32_bf16(false, a, false, wf0[i],
                                                     (short)0, acc0, false, false);
      acc1 = __builtin_amdgcn_wmma_f32_16x16x32_bf16(false, a, false, wf1[i],
                                                     (short)0, acc1, false, false);
    }
    __syncthreads();   // all reads of hl done before overwrite

    // epilogue: h = tanh(acc + x_t + (b_ih+b_hh)); write bf16 back to LDS
#pragma unroll
    for (int r = 0; r < 8; ++r) {
      const int M = r + hiHalf * 8;
      const float* xrow = WihT + (size_t)xi[r] * H_SZ;
      hl[M * HPAD + n0] = (__bf16)FAST_TANH(acc0[r] + xrow[n0] + bias0);
      hl[M * HPAD + n1] = (__bf16)FAST_TANH(acc1[r] + xrow[n1] + bias1);
    }
    __syncthreads();
  }

  // spill final h state (bf16) for the FC kernel
  for (int i = tid; i < 16 * H_SZ; i += 512) {
    int r = i >> 9, c = i & 511;
    hlast[(size_t)(b0 + r) * H_SZ + c] = hl[r * HPAD + c];
  }
}

// ---------------- FC: (128,512)bf16 x (512,4096) -> (128,4096)f32 ----
__global__ __launch_bounds__(256) void fc_kernel(
    const __bf16* __restrict__ hlast, const __bf16* __restrict__ wfc,
    const float* __restrict__ bfc, float* __restrict__ out) {
  const int tid  = threadIdx.x;
  const int lane = tid & 31;
  const int g    = blockIdx.x * 8 + (tid >> 5);  // global wave id 0..2047
  const int mt   = g & 7;                        // batch tile 0..7
  const int nt   = g >> 3;                       // output tile 0..255
  const int Mlo    = lane & 15;
  const int hiHalf = (lane >> 4) & 1;

  const __bf16* arow = hlast + (size_t)(mt * 16 + Mlo) * H_SZ;
  const __bf16* brow = wfc   + (size_t)(nt * 16 + Mlo) * H_SZ;  // B[k][n]=W_fc[n][k]
  const int aCol = hiHalf ? 16 : 0;
  const int bK   = hiHalf ? 16 : 0;

  v8f acc = {0,0,0,0,0,0,0,0};
#pragma unroll
  for (int kb = 0; kb < 16; ++kb) {
    const __bf16* ap = arow + kb * 32 + aCol;
    v16bf a = join16(*(const v8bf*)(ap), *(const v8bf*)(ap + 8));
    const __bf16* bp = brow + kb * 32 + bK;
    v16bf b = join16(*(const v8bf*)(bp), *(const v8bf*)(bp + 8));
    acc = __builtin_amdgcn_wmma_f32_16x16x32_bf16(false, a, false, b,
                                                  (short)0, acc, false, false);
  }

  const int n = nt * 16 + Mlo;
  const float bias = bfc[n];
#pragma unroll
  for (int r = 0; r < 8; ++r) {
    const int M = mt * 16 + r + hiHalf * 8;
    out[(size_t)M * OUT_SZ + n] = acc[r] + bias;
  }
}

extern "C" void kernel_launch(void* const* d_in, const int* in_sizes, int n_in,
                              void* d_out, int out_size, void* d_ws, size_t ws_size,
                              hipStream_t stream) {
  (void)in_sizes; (void)n_in; (void)out_size; (void)ws_size;
  const int*   x    = (const int*)  d_in[0];
  const float* Wih  = (const float*)d_in[1];
  const float* bih  = (const float*)d_in[2];
  const float* Whh  = (const float*)d_in[3];
  const float* bhh  = (const float*)d_in[4];
  const float* Wfc  = (const float*)d_in[5];
  const float* bfc  = (const float*)d_in[6];
  float*       out  = (float*)d_out;

  char* ws = (char*)d_ws;
  float*  WihT   = (float*) (ws + OFF_WIHT);
  __bf16* whh_bf = (__bf16*)(ws + OFF_WHH);
  __bf16* wfc_bf = (__bf16*)(ws + OFF_WFC);
  float*  biasc  = (float*) (ws + OFF_BIAS);
  __bf16* hlast  = (__bf16*)(ws + OFF_HLAST);

  prep_kernel<<<(IN_SZ * H_SZ) / 256, 256, 0, stream>>>(
      Wih, bih, Whh, bhh, Wfc, WihT, whh_bf, wfc_bf, biasc);
  rnn_rec_kernel<<<8, 512, 0, stream>>>(x, WihT, whh_bf, biasc, hlast);
  fc_kernel<<<256, 256, 0, stream>>>(hlast, wfc_bf, bfc, out);
}
